// CausalMultiHypothesisGraphTransformerLayer_3204045603774
// MI455X (gfx1250) — matrix operand order repaired and checked by reading, not compiled, not run on hardware
//
#include <hip/hip_runtime.h>
#include <math.h>

// ---------------------------------------------------------------------------
// MI455X / gfx1250 implementation. All dense algebra runs through
// v_wmma_f32_16x16x32_bf16 (bf16 inputs, f32 accumulate); GEMM K-tiles are
// staged with CDNA5 async global->LDS copies (ASYNCcnt) and double-buffered
// LDS so fetch overlaps WMMA issue. Graph ops (GCN aggregate, GATv2 softmax)
// are VALU/LDS kernels in f32. Working set per stage <= 75MB -> L2 resident.
// ---------------------------------------------------------------------------

typedef __bf16 bf16;
typedef __attribute__((ext_vector_type(16))) __bf16 bf16x16;
typedef __attribute__((ext_vector_type(8)))  __bf16 bf16x8;
typedef __attribute__((ext_vector_type(8)))  float  f32x8;

#define NN   2304   // grid nodes (48*48)
#define BS   4      // batch
#define NB   9216   // NN*BS rows
#define EMB  256
#define HEADS 8
#define HE   2048   // HEADS*EMB

// ======================= graph preparation ================================

__global__ void k_zero_int(int* p, int n) {
    int i = blockIdx.x * 256 + threadIdx.x;
    if (i < n) p[i] = 0;
}

__global__ void k_deg_count(const int* __restrict__ dst, int ne, int* __restrict__ cnt) {
    int e = blockIdx.x * 256 + threadIdx.x;
    if (e < ne) atomicAdd(&cnt[dst[e]], 1);
}

__global__ void k_dinv(const int* __restrict__ cnt, float* __restrict__ dinv, int n) {
    int i = blockIdx.x * 256 + threadIdx.x;
    if (i < n) dinv[i] = rsqrtf((float)cnt[i] + 1.0f);
}

__global__ void k_scan(const int* __restrict__ cnt, int* __restrict__ off, int n) {
    if (blockIdx.x == 0 && threadIdx.x == 0) {
        int a = 0;
        for (int i = 0; i < n; ++i) { off[i] = a; a += cnt[i] + 1; } // +1 self loop
        off[n] = a;
    }
}

// Deterministic CSR fill: per-dst edges in edge-id order, self-loop last.
// Self weight = dinv^2 = 1/deg, exactly the reference's h*(1/deg) term.
__global__ void k_csr_fill(const int* __restrict__ src, const int* __restrict__ dst, int ne,
                           const float* __restrict__ dinv, const int* __restrict__ off,
                           int* __restrict__ csrc, float* __restrict__ cw) {
    int n = blockIdx.x * 256 + threadIdx.x;
    if (n >= NN) return;
    int p = off[n];
    float dn = dinv[n];
    for (int e = 0; e < ne; ++e) {
        if (dst[e] == n) { csrc[p] = src[e]; cw[p] = dinv[src[e]] * dn; ++p; }
    }
    csrc[p] = n; cw[p] = dn * dn;
}

// ======================= weight / input conversion ========================

// WT[n*K+k] = (bf16) W[k*Nc+n]  (transpose so GEMM B-tiles stream along K)
__global__ void k_wT(const float* __restrict__ W, bf16* __restrict__ WT, int K, int Nc) {
    int i = blockIdx.x * 256 + threadIdx.x;
    if (i >= K * Nc) return;
    int n = i / K, k = i - n * K;
    WT[i] = (bf16)W[(size_t)k * Nc + n];
}

// xs[(n*B+b)*768 + c] = {x1,x2,x3}[b][c%256][n]   (gather + bf16)
__global__ void k_build_xs(const float* __restrict__ x1, const float* __restrict__ x2,
                           const float* __restrict__ x3, bf16* __restrict__ xs, int total) {
    int i = blockIdx.x * 256 + threadIdx.x;
    if (i >= total) return;
    int c = i % 768; int r = i / 768; int b = r & 3; int n = r >> 2;
    const float* s = (c < 256) ? x1 : ((c < 512) ? x2 : x3);
    int cc = c & 255;
    xs[i] = (bf16)s[((size_t)b * EMB + cc) * NN + n];
}

// ======================= WMMA bf16 GEMM ===================================
// C[M,Nc] = A[M,K] @ W[K,Nc] (+bias, opt ReLU). A bf16 row-major, WT bf16 [Nc][K].
// Block: 256 thr = 8 waves; tile 128(M) x 64(N); K-step 32; double-buffered
// LDS filled by global_load_async_to_lds_b128 (ASYNCcnt, no VGPR staging).

union FragU { bf16x16 v; bf16x8 h[2]; };

// CDNA5 async copy global->LDS, 16B per lane. The LDS operand is the
// wave-relative LDS byte address; ISA 10.2: flat LDS aperture uses addr[31:0]
// directly as the LDS address, so truncating the generic shared pointer works.
__device__ __forceinline__ void async_ld16(const bf16* lds_p, const bf16* gp) {
    unsigned loff = (unsigned)(size_t)lds_p;
    unsigned long long ga = (unsigned long long)(size_t)gp;
    asm volatile("global_load_async_to_lds_b128 %0, %1, off"
                 :: "v"(loff), "v"(ga) : "memory");
}

template <int ACT>
__global__ __launch_bounds__(256) void gemm_bf16(
    const bf16* __restrict__ A, const bf16* __restrict__ WT,
    const float* __restrict__ bias, float* __restrict__ outF,
    bf16* __restrict__ outB, int M, int Nc, int K) {

    __shared__ bf16 lA[2 * 128 * 40];   // 32 K + 8 pad per row, x2 buffers
    __shared__ bf16 lB[2 * 64 * 40];

    const int m0   = blockIdx.x * 128;
    const int n0   = blockIdx.y * 64;
    const int tid  = threadIdx.x;
    const int wave = tid >> 5;
    const int lane = tid & 31;
    const int lh   = lane & 15;
    const int kA   = (lane < 16) ? 0 : 16;   // A-frag K base per lane-half (contiguous K order)
    const int kB   = (lane < 16) ? 0 : 16;   // B-frag K base per lane-half (contiguous K order)

    f32x8 acc[4];
    #pragma unroll
    for (int t = 0; t < 4; ++t)
        #pragma unroll
        for (int j = 0; j < 8; ++j) acc[t][j] = 0.0f;

    // issue one tile: 2 A-chunks + 1 B-chunk per thread = 3 async ops / wave
    auto stage = [&](int buf, int k0) {
        int c = tid;
        #pragma unroll
        for (int it = 0; it < 2; ++it, c += 256) {
            int row = c >> 2, q = c & 3;
            const bf16* gp = A + (size_t)(m0 + row) * K + k0 + q * 8;
            if (it == 0 && k0 + 32 < K) __builtin_prefetch(gp + 32, 0, 3);
            async_ld16(lA + buf * (128 * 40) + row * 40 + q * 8, gp);
        }
        int n = tid >> 2, q = tid & 3;
        async_ld16(lB + buf * (64 * 40) + n * 40 + q * 8,
                   WT + (size_t)(n0 + n) * K + k0 + q * 8);
    };

    const int steps = K >> 5;
    stage(0, 0);
    for (int i = 0; i < steps; ++i) {
        const int p = i & 1;
        if (i + 1 < steps) {
            stage(p ^ 1, (i + 1) * 32);                      // prefetch next tile
            asm volatile("s_wait_asynccnt 0x3" ::: "memory"); // current 3 retired
        } else {
            asm volatile("s_wait_asynccnt 0x0" ::: "memory");
        }
        __syncthreads();   // all waves' async writes to buffer p visible

        const bf16* Ab = lA + p * (128 * 40);
        const bf16* Bb = lB + p * (64 * 40);

        // A fragment (16x32): elements 0..15 = K kA..kA+15
        FragU af;
        {
            const bf16* ap = Ab + (wave * 16 + lh) * 40 + kA;
            af.h[0] = *(const bf16x8*)(ap);
            af.h[1] = *(const bf16x8*)(ap + 8);
        }
        #pragma unroll
        for (int t = 0; t < 4; ++t) {
            // B fragment (32x16): N = lh, elements 0..15 = K kB..kB+15
            FragU bf;
            const bf16* bp = Bb + (t * 16 + lh) * 40 + kB;
            bf.h[0] = *(const bf16x8*)(bp);
            bf.h[1] = *(const bf16x8*)(bp + 8);
            acc[t] = __builtin_amdgcn_wmma_f32_16x16x32_bf16(
                false, af.v, false, bf.v, (short)0, acc[t], false, false);
        }
        __syncthreads();   // buffer p free for reuse at step i+2
    }

    // C/D map: lanes<16 -> M = r, N = lh ; lanes>=16 -> M = 8+r, N = lh
    const int mbase = m0 + wave * 16 + ((lane < 16) ? 0 : 8);
    #pragma unroll
    for (int t = 0; t < 4; ++t) {
        int col = n0 + t * 16 + lh;
        float bv = bias ? bias[col] : 0.0f;
        #pragma unroll
        for (int r = 0; r < 8; ++r) {
            float v = acc[t][r] + bv;
            if (ACT == 1) v = fmaxf(v, 0.0f);
            size_t idx = (size_t)(mbase + r) * Nc + col;
            if (outF) outF[idx] = v;
            if (outB) outB[idx] = (bf16)v;
        }
    }
}

// ======================= elementwise / fused kernels ======================

__global__ void k_mask(const float* __restrict__ node, const float* __restrict__ z,
                       bf16* __restrict__ xconf, bf16* __restrict__ xadj, int total) {
    int i = blockIdx.x * 256 + threadIdx.x;
    if (i >= total) return;
    float s = 1.0f / (1.0f + expf(-z[i]));
    float nv = node[i];
    xconf[i] = (bf16)(s * nv);           // sigmoid(z)*node
    xadj[i]  = (bf16)((1.0f - s) * nv);  // sigmoid(-z)*node
}

__global__ void k_add3(const float* __restrict__ a, const float* __restrict__ c,
                       const float* __restrict__ iv, bf16* __restrict__ out, int total) {
    int i = blockIdx.x * 256 + threadIdx.x;
    if (i < total) out[i] = (bf16)(a[i] + c[i] + iv[i]);
}

__global__ void k_add2(const float* __restrict__ a, const float* __restrict__ c,
                       float* __restrict__ oF, bf16* __restrict__ oB, int total) {
    int i = blockIdx.x * 256 + threadIdx.x;
    if (i >= total) return;
    float v = a[i] + c[i];
    oF[i] = v; oB[i] = (bf16)v;
}

__device__ __forceinline__ float blockSum256(float v, float* sh) {
    int t = threadIdx.x;
    sh[t] = v; __syncthreads();
    #pragma unroll
    for (int s = 128; s >= 1; s >>= 1) {
        if (t < s) sh[t] += sh[t + s];
        __syncthreads();
    }
    float r = sh[0]; __syncthreads();
    return r;
}

// GCN aggregate (CSR, includes self with w=1/deg) + bias + LayerNorm.
__global__ __launch_bounds__(256) void k_gcn_agg_ln(
    const float* __restrict__ h, const int* __restrict__ off,
    const int* __restrict__ csrc, const float* __restrict__ cw,
    const float* __restrict__ bias, const float* __restrict__ gam,
    const float* __restrict__ bet, float* __restrict__ outF, bf16* __restrict__ outB) {
    __shared__ float sh[256];
    int row = blockIdx.x;            // n*BS + b
    int n = row >> 2, b = row & 3;
    int c = threadIdx.x;
    float acc = bias[c];
    int s1 = off[n + 1];
    for (int s = off[n]; s < s1; ++s)
        acc += cw[s] * h[((size_t)csrc[s] * BS + b) * EMB + c];
    float mean = blockSum256(acc, sh) * (1.0f / EMB);
    float d = acc - mean;
    float var = blockSum256(d * d, sh) * (1.0f / EMB);
    float y = d * rsqrtf(var + 1e-5f) * gam[c] + bet[c];
    size_t idx = (size_t)row * EMB + c;
    outF[idx] = y; outB[idx] = (bf16)y;
}

// GATv2 aggregate: block per (n,b); wave = head, lane = 8-dim slice.
// Online softmax over the node's CSR in-edges; LDS head-mean; writes [B,E,N].
__global__ __launch_bounds__(256) void k_gat_agg(
    const float* __restrict__ xl, const float* __restrict__ xr,
    const int* __restrict__ off, const int* __restrict__ csrc,
    const float* __restrict__ att, const float* __restrict__ bias,
    float* __restrict__ out) {
    __shared__ float buf[HEADS * EMB];
    int row = blockIdx.x;
    int n = row >> 2, b = row & 3;
    int h = threadIdx.x >> 5;
    int lane = threadIdx.x & 31;
    int d0 = lane * 8;

    const float* xrp  = xr + (size_t)row * HE + h * EMB + d0;
    const float* attp = att + h * EMB + d0;
    float xrv[8], attv[8], acc[8];
    #pragma unroll
    for (int j = 0; j < 8; ++j) { xrv[j] = xrp[j]; attv[j] = attp[j]; acc[j] = 0.0f; }

    float m = -INFINITY, den = 0.0f;
    int s1 = off[n + 1];
    for (int s = off[n]; s < s1; ++s) {
        const float* xlp = xl + ((size_t)csrc[s] * BS + b) * HE + h * EMB + d0;
        float xlv[8], part = 0.0f;
        #pragma unroll
        for (int j = 0; j < 8; ++j) {
            xlv[j] = xlp[j];
            float e = xlv[j] + xrv[j];
            e = (e > 0.0f) ? e : 0.2f * e;   // leaky_relu 0.2
            part += attv[j] * e;
        }
        #pragma unroll
        for (int o = 16; o > 0; o >>= 1) part += __shfl_xor(part, o, 32);
        float logit = part;
        float nm = fmaxf(m, logit);
        float sc = expf(m - nm);
        float p  = expf(logit - nm);
        den = den * sc + p;
        #pragma unroll
        for (int j = 0; j < 8; ++j) acc[j] = acc[j] * sc + p * xlv[j];
        m = nm;
    }
    float inv = 1.0f / den;
    #pragma unroll
    for (int j = 0; j < 8; ++j) buf[h * EMB + d0 + j] = acc[j] * inv;
    __syncthreads();

    int d = threadIdx.x;
    float v = 0.0f;
    #pragma unroll
    for (int hh = 0; hh < HEADS; ++hh) v += buf[hh * EMB + d];
    v = v * (1.0f / HEADS) + bias[d];
    out[((size_t)b * EMB + d) * NN + n] = v;   // [B,E,H,W]
}

// residual + LayerNorm + transpose to [B,E,N]
__global__ __launch_bounds__(256) void k_add_ln_out(
    const float* __restrict__ orig, const float* __restrict__ v,
    const float* __restrict__ gam, const float* __restrict__ bet,
    float* __restrict__ out) {
    __shared__ float sh[256];
    int row = blockIdx.x;
    int n = row >> 2, b = row & 3;
    int c = threadIdx.x;
    float x = orig[(size_t)row * EMB + c] + v[(size_t)row * EMB + c];
    float mean = blockSum256(x, sh) * (1.0f / EMB);
    float d = x - mean;
    float var = blockSum256(d * d, sh) * (1.0f / EMB);
    float y = d * rsqrtf(var + 1e-5f) * gam[c] + bet[c];
    out[((size_t)b * EMB + c) * NN + n] = y;
}

// ======================= launch =============================================

extern "C" void kernel_launch(void* const* d_in, const int* in_sizes, int n_in,
                              void* d_out, int out_size, void* d_ws, size_t ws_size,
                              hipStream_t stream) {
    const float* x1 = (const float*)d_in[0];
    const float* x2 = (const float*)d_in[1];
    const float* x3 = (const float*)d_in[2];
    const int*   ei = (const int*)d_in[3];
    const int    Ne = in_sizes[3] / 2;
    const int*   esrc = ei;
    const int*   edst = ei + Ne;

    const float* projW = (const float*)d_in[4];
    const float* projB = (const float*)d_in[5];
    const float* mgW1  = (const float*)d_in[6];
    const float* mgB1  = (const float*)d_in[7];
    const float* mgW2  = (const float*)d_in[8];
    const float* mgB2  = (const float*)d_in[9];
    const float* gAW   = (const float*)d_in[10];
    const float* gAB   = (const float*)d_in[11];
    const float* gCW   = (const float*)d_in[12];
    const float* gCB   = (const float*)d_in[13];
    const float* ln0g  = (const float*)d_in[14];
    const float* ln0b  = (const float*)d_in[15];
    const float* ln1g  = (const float*)d_in[16];
    const float* ln1b  = (const float*)d_in[17];
    const float* ln2g  = (const float*)d_in[18];
    const float* ln2b  = (const float*)d_in[19];
    const float *civW1[3], *civB1[3], *civW2[3], *civB2[3];
    for (int i = 0; i < 3; ++i) {
        civW1[i] = (const float*)d_in[20 + 4 * i];
        civB1[i] = (const float*)d_in[21 + 4 * i];
        civW2[i] = (const float*)d_in[22 + 4 * i];
        civB2[i] = (const float*)d_in[23 + 4 * i];
    }
    const float *gatWl[3], *gatWr[3], *gatAtt[3], *gatB[3];
    for (int i = 0; i < 3; ++i) {
        gatWl[i]  = (const float*)d_in[32 + 4 * i];
        gatWr[i]  = (const float*)d_in[33 + 4 * i];
        gatAtt[i] = (const float*)d_in[34 + 4 * i];
        gatB[i]   = (const float*)d_in[35 + 4 * i];
    }
    const float* fW1 = (const float*)d_in[44];
    const float* fB1 = (const float*)d_in[45];
    const float* fW2 = (const float*)d_in[46];
    const float* fB2 = (const float*)d_in[47];

    // ---- workspace carve (bump allocator, 256B aligned; ~310 MB total) ----
    char* wsb = (char*)d_ws;
    size_t ofs = 0;
    auto alloc = [&](size_t bytes) -> void* {
        void* p = wsb + ofs;
        ofs = (ofs + bytes + 255) & ~(size_t)255;
        return p;
    };
    int*   cnt  = (int*)alloc(NN * 4);
    float* dinv = (float*)alloc(NN * 4);
    int*   off  = (int*)alloc((NN + 1) * 4);
    int*   csrc = (int*)alloc((size_t)(Ne + NN) * 4);
    float* cw   = (float*)alloc((size_t)(Ne + NN) * 4);

    bf16* projT = (bf16*)alloc((size_t)768 * 256 * 2);
    bf16* mg1T  = (bf16*)alloc((size_t)256 * 128 * 2);
    bf16* mg2T  = (bf16*)alloc((size_t)128 * 256 * 2);
    bf16* gAT   = (bf16*)alloc((size_t)256 * 256 * 2);
    bf16* gCT   = (bf16*)alloc((size_t)256 * 256 * 2);
    bf16 *c1T[3], *c2T[3], *wlT[3], *wrT[3];
    for (int i = 0; i < 3; ++i) {
        c1T[i] = (bf16*)alloc((size_t)256 * 128 * 2);
        c2T[i] = (bf16*)alloc((size_t)128 * 256 * 2);
        wlT[i] = (bf16*)alloc((size_t)256 * 2048 * 2);
        wrT[i] = (bf16*)alloc((size_t)256 * 2048 * 2);
    }
    bf16* f1T = (bf16*)alloc((size_t)256 * 512 * 2);
    bf16* f2T = (bf16*)alloc((size_t)512 * 256 * 2);

    bf16*  xs    = (bf16*)alloc((size_t)NB * 768 * 2);
    float* nodeF = (float*)alloc((size_t)NB * EMB * 4);
    bf16*  nodeB = (bf16*)alloc((size_t)NB * EMB * 2);
    bf16*  h1B   = (bf16*)alloc((size_t)NB * 128 * 2);
    float* zF    = (float*)alloc((size_t)NB * EMB * 4);
    bf16*  xconf = (bf16*)alloc((size_t)NB * EMB * 2);
    bf16*  xadj  = (bf16*)alloc((size_t)NB * EMB * 2);
    float* hbuf  = (float*)alloc((size_t)NB * EMB * 4);
    float* adjF  = (float*)alloc((size_t)NB * EMB * 4);
    bf16*  adjB  = (bf16*)alloc((size_t)NB * EMB * 2);
    float* confF = (float*)alloc((size_t)NB * EMB * 4);
    bf16*  confB = (bf16*)alloc((size_t)NB * EMB * 2);
    bf16*  tB    = (bf16*)alloc((size_t)NB * 128 * 2);
    float* ivF   = (float*)alloc((size_t)NB * EMB * 4);
    bf16*  featB = (bf16*)alloc((size_t)NB * EMB * 2);
    float* xlF   = (float*)alloc((size_t)NB * HE * 4);
    float* xrF   = (float*)alloc((size_t)NB * HE * 4);
    float* origF = (float*)alloc((size_t)NB * EMB * 4);
    bf16*  origB = (bf16*)alloc((size_t)NB * EMB * 2);
    bf16*  uB    = (bf16*)alloc((size_t)NB * 512 * 2);
    float* vF    = (float*)alloc((size_t)NB * EMB * 4);

    float* outP = (float*)d_out;
    const size_t OUT1 = (size_t)BS * EMB * NN;

    // ---- graph prep ----
    k_zero_int<<<(NN + 255) / 256, 256, 0, stream>>>(cnt, NN);
    k_deg_count<<<(Ne + 255) / 256, 256, 0, stream>>>(edst, Ne, cnt);
    k_dinv<<<(NN + 255) / 256, 256, 0, stream>>>(cnt, dinv, NN);
    k_scan<<<1, 1, 0, stream>>>(cnt, off, NN);
    k_csr_fill<<<(NN + 255) / 256, 256, 0, stream>>>(esrc, edst, Ne, dinv, off, csrc, cw);

    // ---- weight transpose/convert ----
    auto cvt = [&](const float* W, bf16* WT, int K, int Nc) {
        int tot = K * Nc;
        k_wT<<<(tot + 255) / 256, 256, 0, stream>>>(W, WT, K, Nc);
    };
    cvt(projW, projT, 768, 256);
    cvt(mgW1, mg1T, 256, 128);  cvt(mgW2, mg2T, 128, 256);
    cvt(gAW, gAT, 256, 256);    cvt(gCW, gCT, 256, 256);
    for (int i = 0; i < 3; ++i) {
        cvt(civW1[i], c1T[i], 256, 128);  cvt(civW2[i], c2T[i], 128, 256);
        cvt(gatWl[i], wlT[i], 256, 2048); cvt(gatWr[i], wrT[i], 256, 2048);
    }
    cvt(fW1, f1T, 256, 512); cvt(fW2, f2T, 512, 256);

    // ---- node features ----
    {
        int tot = NB * 768;
        k_build_xs<<<(tot + 255) / 256, 256, 0, stream>>>(x1, x2, x3, xs, tot);
    }

    auto gemm = [&](const bf16* A, const bf16* WT, const float* bias,
                    float* oF, bf16* oB, int M, int Nc, int K, int act) {
        dim3 g(M / 128, Nc / 64);
        if (act) gemm_bf16<1><<<g, 256, 0, stream>>>(A, WT, bias, oF, oB, M, Nc, K);
        else     gemm_bf16<0><<<g, 256, 0, stream>>>(A, WT, bias, oF, oB, M, Nc, K);
    };
    const int EW = NB * EMB;

    // projection
    gemm(xs, projT, projB, nodeF, nodeB, NB, 256, 768, 0);
    // dynamic mask disentangle
    gemm(nodeB, mg1T, mgB1, nullptr, h1B, NB, 128, 256, 1);
    gemm(h1B, mg2T, mgB2, zF, nullptr, NB, 256, 128, 0);
    k_mask<<<(EW + 255) / 256, 256, 0, stream>>>(nodeF, zF, xconf, xadj, EW);
    // two GCN branches (+LN fused)
    gemm(xadj, gAT, nullptr, hbuf, nullptr, NB, 256, 256, 0);
    k_gcn_agg_ln<<<NB, 256, 0, stream>>>(hbuf, off, csrc, cw, gAB, ln0g, ln0b, adjF, adjB);
    gemm(xconf, gCT, nullptr, hbuf, nullptr, NB, 256, 256, 0);
    k_gcn_agg_ln<<<NB, 256, 0, stream>>>(hbuf, off, csrc, cw, gCB, ln1g, ln1b, confF, confB);
    // hypotheses
    for (int i = 0; i < 3; ++i) {
        gemm(confB, c1T[i], civB1[i], nullptr, tB, NB, 128, 256, 1);
        gemm(tB, c2T[i], civB2[i], ivF, nullptr, NB, 256, 128, 0);
        k_add3<<<(EW + 255) / 256, 256, 0, stream>>>(adjF, confF, ivF, featB, EW);
        gemm(featB, wlT[i], nullptr, xlF, nullptr, NB, 2048, 256, 0);
        gemm(featB, wrT[i], nullptr, xrF, nullptr, NB, 2048, 256, 0);
        k_gat_agg<<<NB, 256, 0, stream>>>(xlF, xrF, off, csrc, gatAtt[i], gatB[i],
                                          outP + (size_t)i * OUT1);
    }
    // FFN + residual LN
    k_add2<<<(EW + 255) / 256, 256, 0, stream>>>(adjF, confF, origF, origB, EW);
    gemm(origB, f1T, fB1, nullptr, uB, NB, 512, 256, 1);
    gemm(uB, f2T, fB2, vF, nullptr, NB, 256, 512, 0);
    k_add_ln_out<<<NB, 256, 0, stream>>>(origF, vF, ln2g, ln2b, outP + 3 * OUT1);
}
